// MLXAttention_89386859364894
// MI455X (gfx1250) — matrix operand order repaired and checked by reading, not compiled, not run on hardware
//
#include <hip/hip_runtime.h>
#include <hip/hip_bf16.h>

// ---------------------------------------------------------------------------
// MI455X (gfx1250, wave32) GQA attention: bf16 WMMA pipeline with
// double-buffered async global->LDS tile staging (ASYNCcnt path).
//   x,W -> bf16 ; q/k/v = GEMM ; flash attention (online softmax) ; o = GEMM
// ---------------------------------------------------------------------------

typedef __attribute__((ext_vector_type(8)))  __bf16 v8bf;
typedef __attribute__((ext_vector_type(16))) __bf16 v16bf;
typedef __attribute__((ext_vector_type(8)))  float  v8f;

union F16x16 { v16bf v; v8bf h[2]; __bf16 e[16]; };

__device__ __forceinline__ __bf16 f2bf(float f) {
  // round-to-nearest-even f32 -> bf16
  unsigned int u = __builtin_bit_cast(unsigned int, f);
  unsigned int lsb = (u >> 16) & 1u;
  u += 0x7fffu + lsb;
  unsigned short h = (unsigned short)(u >> 16);
  return __builtin_bit_cast(__bf16, h);
}

__device__ __forceinline__ v8f vzero8() {
  v8f z;
#pragma unroll
  for (int i = 0; i < 8; ++i) z[i] = 0.0f;
  return z;
}

// --- CDNA5 async global->LDS (per-lane 16B), tracked by ASYNCcnt -----------
__device__ __forceinline__ unsigned int lds_off(const void* p) {
  // generic 'shared' address: low 32 bits are the LDS byte offset
  return (unsigned int)(unsigned long long)p;
}

__device__ __forceinline__ void async_load_b128(unsigned int lds_byte_off,
                                                const void* gptr) {
  asm volatile("global_load_async_to_lds_b128 %0, %1, off"
               :
               : "v"(lds_byte_off), "v"((unsigned long long)gptr)
               : "memory");
}

__device__ __forceinline__ void wait_async0() {
  asm volatile("s_wait_asynccnt 0x0" ::: "memory");
}

// A-matrix fragment (16x32 bf16, MxK): lane holds row m=lane&15,
// K = kb..kb+15 with kb = 16*(lane>>4).
__device__ __forceinline__ v16bf load_a_frag(const __bf16* base, int lda, int lane) {
  const int m  = lane & 15;
  const int kb = (lane >> 4) << 4;
  const __bf16* p = base + m * lda + kb;
  F16x16 f;
  f.h[0] = *(const v8bf*)(p);
  f.h[1] = *(const v8bf*)(p + 8);
  return f.v;
}

// B-matrix fragment (32x16 bf16, KxN) from a tile stored [n][k] (K contiguous):
// lane holds col n=lane&15, K = kb..kb+15 with kb = 16*(lane>>4).
__device__ __forceinline__ v16bf load_b_frag_rowk(const __bf16* base, int ldn, int lane) {
  const int n  = lane & 15;
  const int kb = (lane >> 4) << 4;
  const __bf16* p = base + n * ldn + kb;
  F16x16 f;
  f.h[0] = *(const v8bf*)(p);
  f.h[1] = *(const v8bf*)(p + 8);
  return f.v;
}

// B-matrix fragment from a tile stored [k][n] (N contiguous) -> strided gather.
__device__ __forceinline__ v16bf load_b_frag_colk(const __bf16* base, int ldk, int lane) {
  const int n  = lane & 15;
  const int kb = (lane >> 4) << 4;
  F16x16 f;
#pragma unroll
  for (int t = 0; t < 16; ++t) f.e[t] = base[(kb + t) * ldk + n];
  return f.v;
}

__device__ __forceinline__ v8f wmma_bf16(v16bf a, v16bf b, v8f c) {
  // D = A(16x32) * B(32x16) + C, f32 accumulate
  return __builtin_amdgcn_wmma_f32_16x16x32_bf16(false, a, false, b, (short)0, c,
                                                 false, false);
}

// ---------------------------------------------------------------------------
// f32 -> bf16 conversion
// ---------------------------------------------------------------------------
__global__ void cvt_f32_bf16(const float* __restrict__ in, __bf16* __restrict__ out,
                             int n) {
  int i = blockIdx.x * blockDim.x + threadIdx.x;
  const int stride = gridDim.x * blockDim.x;
  for (; i < n; i += stride) out[i] = f2bf(in[i]);
}

// ---------------------------------------------------------------------------
// TN GEMM:  C[M,N] = A[M,K] * W[N,K]^T + bias[N]
// block tile 128x128, BK = 32 (= one WMMA K step), 256 threads = 8 waves,
// each wave computes a 64x32 sub-tile (4x2 WMMA tiles).
// Double-buffered async global->LDS staging: tile i+1 streams in during
// tile i's WMMA work; one s_wait_asynccnt + two barriers per K-step.
// ---------------------------------------------------------------------------
template <typename OutT>
__global__ __launch_bounds__(256) void gemm_tn_bf16(
    const __bf16* __restrict__ A, const __bf16* __restrict__ W,
    const float* __restrict__ bias, OutT* __restrict__ C,
    int M, int N, int K) {
  __shared__ __align__(32) __bf16 As[2][128 * 32];
  __shared__ __align__(32) __bf16 Ws[2][128 * 32];

  const int tid  = threadIdx.x;
  const int lane = tid & 31;
  const int wave = tid >> 5;
  const int wm   = wave >> 2;   // 0..1
  const int wn   = wave & 3;    // 0..3
  const int nBase = blockIdx.x * 128;
  const int mBase = blockIdx.y * 128;

  const int lrow = tid >> 1;        // 0..127
  const int lk   = (tid & 1) * 16;  // 0 or 16

  const __bf16* gA = A + (size_t)(mBase + lrow) * K + lk;
  const __bf16* gW = W + (size_t)(nBase + lrow) * K + lk;

  auto issue_tile = [&](int i, int buf) {
    const __bf16* sa = gA + i * 32;
    async_load_b128(lds_off(&As[buf][lrow * 32 + lk]),     sa);
    async_load_b128(lds_off(&As[buf][lrow * 32 + lk + 8]), sa + 8);
    const __bf16* sw = gW + i * 32;
    async_load_b128(lds_off(&Ws[buf][lrow * 32 + lk]),     sw);
    async_load_b128(lds_off(&Ws[buf][lrow * 32 + lk + 8]), sw + 8);
  };

  v8f acc[4][2];
#pragma unroll
  for (int i = 0; i < 4; ++i)
#pragma unroll
    for (int j = 0; j < 2; ++j) acc[i][j] = vzero8();

  const int nk = K >> 5;
  issue_tile(0, 0);

  for (int i = 0; i < nk; ++i) {
    const int buf = i & 1;
    wait_async0();       // tile i finished landing in LDS (this wave's ops)
    __syncthreads();     // all waves landed; prior reads of buf^1 complete
    if (i + 1 < nk) issue_tile(i + 1, buf ^ 1);

    v16bf bf[2];
#pragma unroll
    for (int nt = 0; nt < 2; ++nt)
      bf[nt] = load_b_frag_rowk(&Ws[buf][(wn * 32 + nt * 16) * 32], 32, lane);
#pragma unroll
    for (int mt = 0; mt < 4; ++mt) {
      v16bf af = load_a_frag(&As[buf][(wm * 64 + mt * 16) * 32], 32, lane);
#pragma unroll
      for (int nt = 0; nt < 2; ++nt)
        acc[mt][nt] = wmma_bf16(af, bf[nt], acc[mt][nt]);
    }
    __syncthreads();     // reads of buf done before its async overwrite next+1
  }

  // epilogue: C-frag rows = 8*(lane>>4)+r, col = lane&15
#pragma unroll
  for (int mt = 0; mt < 4; ++mt) {
#pragma unroll
    for (int nt = 0; nt < 2; ++nt) {
      const int gn = nBase + wn * 32 + nt * 16 + (lane & 15);
      const float bv = bias[gn];
#pragma unroll
      for (int r = 0; r < 8; ++r) {
        const int gm = mBase + wm * 64 + mt * 16 + ((lane >> 4) << 3) + r;
        const float v = acc[mt][nt][r] + bv;
        if constexpr (sizeof(OutT) == 2)
          C[(size_t)gm * N + gn] = (OutT)f2bf(v);
        else
          C[(size_t)gm * N + gn] = v;
      }
    }
  }
}

// ---------------------------------------------------------------------------
// Flash attention (causal, GQA head h uses kv head h % KV).
// Block: one (b, h, 128 query rows). 8 waves x 16 q-rows each.
// K/V staged via double-buffered async global->LDS in 32-key tiles.
// ---------------------------------------------------------------------------
__global__ __launch_bounds__(256) void flash_attn_gqa(
    const __bf16* __restrict__ Qg, const __bf16* __restrict__ Kg,
    const __bf16* __restrict__ Vg, __bf16* __restrict__ Og) {
  constexpr int Sx = 2048, HIDx = 4096, Hx = 32, KVx = 8, Dx = 128, KVDx = 1024;
  constexpr int QB = 128, KT = 32, LDK = 136;  // pad 128->136: dodge bank conflicts
  const float scale = 0.08838834764831845f;    // 1/sqrt(128)

  __shared__ __align__(32) __bf16 Ks[2][KT * LDK];
  __shared__ __align__(32) __bf16 Vs[2][KT * LDK];
  __shared__ __align__(32) __bf16 Ps[8 * 16 * 32];  // per-wave P re-layout stash

  const int tid = threadIdx.x, lane = tid & 31, wave = tid >> 5;
  const int qtiles = Sx / QB;  // 16
  const int qt = blockIdx.x % qtiles;
  const int h  = (blockIdx.x / qtiles) % Hx;
  const int b  = blockIdx.x / (qtiles * Hx);
  const int kvh = h % KVx;
  const int q0 = qt * QB + wave * 16;

  // Q fragments stay in VGPRs for the whole kernel (4 K-chunks of 32)
  v16bf aq[4];
  const __bf16* qbase = Qg + ((size_t)(b * Sx + q0) * HIDx + h * Dx);
#pragma unroll
  for (int c = 0; c < 4; ++c) aq[c] = load_a_frag(qbase + 32 * c, HIDx, lane);

  v8f o[8];
#pragma unroll
  for (int i = 0; i < 8; ++i) o[i] = vzero8();
  float mrow[8], lrow[8];
#pragma unroll
  for (int r = 0; r < 8; ++r) { mrow[r] = -1e30f; lrow[r] = 0.0f; }

  const int nkt    = qt * (QB / KT) + (QB / KT);  // causal: stop at diagonal
  const int keyrow = tid >> 3;                    // 0..31
  const int dseg   = (tid & 7) * 16;              // 0..112
  const int colb   = lane & 15;
  const int rowb   = (lane >> 4) << 3;

  const __bf16* gK = Kg + ((size_t)(b * Sx + keyrow) * KVDx + kvh * Dx + dseg);
  const __bf16* gV = Vg + ((size_t)(b * Sx + keyrow) * KVDx + kvh * Dx + dseg);

  auto issue_kv = [&](int kt, int buf) {
    const size_t off = (size_t)kt * KT * KVDx;
    async_load_b128(lds_off(&Ks[buf][keyrow * LDK + dseg]),     gK + off);
    async_load_b128(lds_off(&Ks[buf][keyrow * LDK + dseg + 8]), gK + off + 8);
    async_load_b128(lds_off(&Vs[buf][keyrow * LDK + dseg]),     gV + off);
    async_load_b128(lds_off(&Vs[buf][keyrow * LDK + dseg + 8]), gV + off + 8);
  };

  issue_kv(0, 0);

  for (int kt = 0; kt < nkt; ++kt) {
    const int kbase = kt * KT;
    const int buf = kt & 1;
    wait_async0();       // K/V tile kt landed (this wave's async ops)
    __syncthreads();     // all waves landed; prior reads of buf^1 complete
    if (kt + 1 < nkt) issue_kv(kt + 1, buf ^ 1);

    // S = Q * K^T  (16x32 per wave: 2 n-tiles x 4 K-chunks = 8 WMMAs)
    v8f sf[2];
    sf[0] = vzero8(); sf[1] = vzero8();
#pragma unroll
    for (int c = 0; c < 4; ++c) {
#pragma unroll
      for (int nt = 0; nt < 2; ++nt) {
        v16bf bk = load_b_frag_rowk(&Ks[buf][(nt * 16) * LDK + 32 * c], LDK, lane);
        sf[nt] = wmma_bf16(aq[c], bk, sf[nt]);
      }
    }

    // online softmax (f32). C-frag: row rowb+r, cols colb / colb+16.
#pragma unroll
    for (int r = 0; r < 8; ++r) {
      const int qg = q0 + rowb + r;
      float s0 = sf[0][r] * scale;
      float s1 = sf[1][r] * scale;
      if (kbase + colb      > qg) s0 = -1e30f;  // causal mask
      if (kbase + colb + 16 > qg) s1 = -1e30f;
      float tmax = fmaxf(s0, s1);
#pragma unroll
      for (int off = 1; off < 16; off <<= 1)
        tmax = fmaxf(tmax, __shfl_xor(tmax, off, 32));
      const float mnew = fmaxf(mrow[r], tmax);
      const float corr = __expf(mrow[r] - mnew);
      const float p0 = __expf(s0 - mnew);
      const float p1 = __expf(s1 - mnew);
      float psum = p0 + p1;
#pragma unroll
      for (int off = 1; off < 16; off <<= 1)
        psum += __shfl_xor(psum, off, 32);
      lrow[r] = lrow[r] * corr + psum;
      mrow[r] = mnew;
#pragma unroll
      for (int dt = 0; dt < 8; ++dt) o[dt][r] *= corr;
      // stash P bf16 in per-wave LDS region (C-layout -> A-layout relayout)
      Ps[wave * 512 + (rowb + r) * 32 + colb]      = f2bf(p0);
      Ps[wave * 512 + (rowb + r) * 32 + colb + 16] = f2bf(p1);
    }

    // O += P(16x32) * V(32x128): 8 WMMAs (LDS ops are in-order within a wave)
    v16bf ap = load_a_frag(&Ps[wave * 512], 32, lane);
#pragma unroll
    for (int dt = 0; dt < 8; ++dt) {
      v16bf bv = load_b_frag_colk(&Vs[buf][dt * 16], LDK, lane);
      o[dt] = wmma_bf16(ap, bv, o[dt]);
    }
  }

  // normalize and write attn output [b, s, h*D + d] as bf16
#pragma unroll
  for (int dt = 0; dt < 8; ++dt) {
#pragma unroll
    for (int r = 0; r < 8; ++r) {
      const int spos = q0 + rowb + r;
      const int col  = h * Dx + dt * 16 + colb;
      Og[(size_t)(b * Sx + spos) * HIDx + col] = f2bf(o[dt][r] / lrow[r]);
    }
  }
}

// ---------------------------------------------------------------------------
// launcher
// ---------------------------------------------------------------------------
extern "C" void kernel_launch(void* const* d_in, const int* in_sizes, int n_in,
                              void* d_out, int out_size, void* d_ws, size_t ws_size,
                              hipStream_t stream) {
  constexpr int Bx = 2, Sx = 2048, HIDx = 4096, Hx = 32, KVDx = 1024;

  const float* x   = (const float*)d_in[0];
  // d_in[1] = additive causal mask: applied analytically in the kernel
  const float* q_w = (const float*)d_in[2];
  const float* q_b = (const float*)d_in[3];
  const float* k_w = (const float*)d_in[4];
  const float* k_b = (const float*)d_in[5];
  const float* v_w = (const float*)d_in[6];
  const float* v_b = (const float*)d_in[7];
  const float* o_w = (const float*)d_in[8];
  const float* o_b = (const float*)d_in[9];
  float* out = (float*)d_out;

  const size_t NX  = (size_t)Bx * Sx * HIDx;  // 16.78M
  const size_t NWQ = (size_t)HIDx * HIDx;     // 16.78M
  const size_t NWK = (size_t)KVDx * HIDx;     // 4.19M
  const size_t NKV = (size_t)Bx * Sx * KVDx;  // 4.19M

  __bf16* p   = (__bf16*)d_ws;
  __bf16* xb  = p; p += NX;
  __bf16* qwb = p; p += NWQ;
  __bf16* kwb = p; p += NWK;
  __bf16* vwb = p; p += NWK;
  __bf16* owb = p; p += NWQ;
  __bf16* qb_ = p; p += NX;
  __bf16* kb_ = p; p += NKV;
  __bf16* vb_ = p; p += NKV;
  __bf16* ab_ = p; p += NX;

  auto cvt = [&](const float* s, __bf16* d, size_t n) {
    int blocks = (int)((n + 255) / 256);
    if (blocks > 8192) blocks = 8192;
    cvt_f32_bf16<<<blocks, 256, 0, stream>>>(s, d, (int)n);
  };
  cvt(x,   xb,  NX);
  cvt(q_w, qwb, NWQ);
  cvt(k_w, kwb, NWK);
  cvt(v_w, vwb, NWK);
  cvt(o_w, owb, NWQ);

  const int M = Bx * Sx;  // 4096
  dim3 blk(256);
  gemm_tn_bf16<__bf16><<<dim3(HIDx / 128, M / 128), blk, 0, stream>>>(
      xb, qwb, q_b, qb_, M, HIDx, HIDx);
  gemm_tn_bf16<__bf16><<<dim3(KVDx / 128, M / 128), blk, 0, stream>>>(
      xb, kwb, k_b, kb_, M, KVDx, HIDx);
  gemm_tn_bf16<__bf16><<<dim3(KVDx / 128, M / 128), blk, 0, stream>>>(
      xb, vwb, v_b, vb_, M, KVDx, HIDx);

  flash_attn_gqa<<<dim3(Bx * Hx * (Sx / 128)), blk, 0, stream>>>(qb_, kb_, vb_, ab_);

  gemm_tn_bf16<float><<<dim3(HIDx / 128, M / 128), blk, 0, stream>>>(
      ab_, owb, o_b, out, M, HIDx, HIDx);
}
